// RealMambaClassifier_11029476016772
// MI455X (gfx1250) — compile-verified, matrix operand order checked by reading, not run on hardware
//
#include <hip/hip_runtime.h>

// ---------------------------------------------------------------------------
// Mamba classifier for MI455X (gfx1250, wave32).
// Big GEMMs: v_wmma_f32_16x16x32_bf16, f32 accumulate. Weight panels are
// async-staged into LDS (global_load_async_to_lds_b128 / s_wait_asynccnt);
// each wave computes a 16 x (NBLK*16) output strip reusing its A fragment.
// NBLK is a template constant so the hot loop has zero guards/branches.
// Elementwise math + the sequential selective scan stay f32.
// ---------------------------------------------------------------------------

#define BATCH   16
#define SEQ     1024
#define NIN     40
#define DM      256
#define DS      16
#define NL      4
#define DC      4
#define DI      512           // EXPAND * DM
#define DTR     16
#define XD      48            // DTR + 2*DS
#define BL      (BATCH*SEQ)   // 16384 tokens

typedef __bf16 bf16;
typedef __attribute__((ext_vector_type(16))) __bf16 v16bf;
typedef __attribute__((ext_vector_type(8)))  __bf16 v8bf;
typedef __attribute__((ext_vector_type(8)))  float  v8f;

__device__ __forceinline__ float siluf(float x)     { return x / (1.f + __expf(-x)); }
__device__ __forceinline__ float softplusf(float x) { return (x > 20.f) ? x : log1pf(__expf(x)); }

// ---------------------------------------------------------------------------
// C[m,n] (+)= sum_k A[m,k] * W[n,k].  A: M x K bf16, W: N x K bf16, C: M x N f32.
// Block = 8 waves covering 128 rows x NBLK*16 cols. The weight panel
// (contiguous, row-major N x K) is async-copied to LDS once per block and
// shared by all 8 waves; each wave does 1 M-tile x NBLK N-tiles. K % 32 == 0,
// (N/16) % NBLK == 0 (exact grid fit -> no guards, EXEC all-ones at WMMA).
// Fragment layouts per CDNA5 ISA 7.12.2.
// ---------------------------------------------------------------------------
template <int NBLK>
__global__ void gemm_bf16_wmma(const bf16* __restrict__ A, const bf16* __restrict__ W,
                               float* __restrict__ C, int M, int N, int K, int accum) {
    extern __shared__ char smem[];
    bf16* sB = (bf16*)smem;

    const int wave = threadIdx.x >> 5;
    const int lane = threadIdx.x & 31;
    const int nT0  = blockIdx.x * NBLK;
    const int mT   = blockIdx.y * 8 + wave;          // exact fit

    // ---- async-stage the shared weight panel (NBLK*16 rows x K bf16) ----
    {
        const int chunks = (NBLK * 16 * K * 2) >> 4;            // 16B chunks
        const unsigned long long gbase =
            (unsigned long long)(const void*)(W + (size_t)nT0 * 16 * K);
        const unsigned ldsbase = __builtin_amdgcn_groupstaticsize(); // dyn-LDS base
        for (int c = threadIdx.x; c < chunks; c += 256) {
            unsigned          lds = ldsbase + (unsigned)c * 16;
            unsigned long long ga = gbase + (unsigned long long)c * 16;
            asm volatile("global_load_async_to_lds_b128 %0, %1, off"
                         :: "v"(lds), "v"(ga) : "memory");
        }
        asm volatile("s_wait_asynccnt 0" ::: "memory");
    }
    __syncthreads();

    const int hs = lane >> 4;   // K half-select (0 or 8)
    const int r  = lane & 15;   // A row / B column within tile

    const bf16* aRow = A + (size_t)(mT * 16 + r) * K + hs * 8;

    v8f acc[NBLK];
    #pragma unroll
    for (int j = 0; j < NBLK; ++j) acc[j] = (v8f){0.f,0.f,0.f,0.f,0.f,0.f,0.f,0.f};

    const size_t cb0 = (size_t)(mT * 16 + hs * 8) * N + r;  // + nT*16 per tile
    if (accum) {
        #pragma unroll
        for (int j = 0; j < NBLK; ++j) {
            const size_t cb = cb0 + (size_t)(nT0 + j) * 16;
            #pragma unroll
            for (int i = 0; i < 8; ++i) acc[j][i] = C[cb + (size_t)i * N];
        }
    }

    for (int k0 = 0; k0 < K; k0 += 32) {
        v8bf a0 = *(const v8bf*)(aRow + k0);                 // K = k0+hs8   ..+7
        v8bf a1 = *(const v8bf*)(aRow + k0 + 16);            // K = k0+16+hs8..+7
        v16bf av = __builtin_shufflevector(a0, a1, 0,1,2,3,4,5,6,7,8,9,10,11,12,13,14,15);
        #pragma unroll
        for (int j = 0; j < NBLK; ++j) {
            const bf16* bRow = sB + (size_t)(j * 16 + r) * K + hs * 8;
            v8bf b0 = *(const v8bf*)(bRow + k0);             // ds_load_b128
            v8bf b1 = *(const v8bf*)(bRow + k0 + 16);
            v16bf bv = __builtin_shufflevector(b0, b1, 0,1,2,3,4,5,6,7,8,9,10,11,12,13,14,15);
            acc[j] = __builtin_amdgcn_wmma_f32_16x16x32_bf16(false, av, false, bv,
                                                             (short)0, acc[j], false, false);
        }
    }

    #pragma unroll
    for (int j = 0; j < NBLK; ++j) {
        const size_t cb = cb0 + (size_t)(nT0 + j) * 16;
        #pragma unroll
        for (int i = 0; i < 8; ++i) C[cb + (size_t)i * N] = acc[j][i];
    }
}

// --------------------------- helper kernels --------------------------------
__global__ void cvt_bf16_kernel(const float* __restrict__ src, bf16* __restrict__ dst, int n) {
    int i = blockIdx.x * blockDim.x + threadIdx.x;
    if (i < n) dst[i] = (bf16)src[i];
}

// dt_proj_w (NL*DI x 16) -> (NL*DI x 32) bf16, zero padded K 16->32
__global__ void dtw_pad_kernel(const float* __restrict__ src, bf16* __restrict__ dst) {
    int i = blockIdx.x * blockDim.x + threadIdx.x;   // NL*DI*32
    if (i >= NL * DI * 32) return;
    int j = i & 31, row = i >> 5;
    dst[i] = (bf16)(j < DTR ? src[row * DTR + j] : 0.f);
}

// h[t,c] = x[t,:40] @ ipw[c,:40] + ipb[c]   (K=40, ~0.6% of FLOPs: scalar f32)
__global__ void inproj_kernel(const float* __restrict__ x, const float* __restrict__ w,
                              const float* __restrict__ b, float* __restrict__ h) {
    int i = blockIdx.x * blockDim.x + threadIdx.x;   // BL*DM
    if (i >= BL * DM) return;
    int c = i & (DM - 1);
    size_t row = (size_t)(i >> 8);
    float s = b[c];
    const float* xp = x + row * NIN;
    const float* wp = w + (size_t)c * NIN;
    #pragma unroll 8
    for (int k = 0; k < NIN; ++k) s += xp[k] * wp[k];
    h[i] = s;
}

// layernorm over DM=256, one wave per row, bf16 output feeding the WMMA GEMM
__global__ void ln_bf16_kernel(const float* __restrict__ h, const float* __restrict__ g,
                               const float* __restrict__ bta, bf16* __restrict__ out) {
    int wave = threadIdx.x >> 5, lane = threadIdx.x & 31;
    size_t row = (size_t)blockIdx.x * 8 + wave;
    const float* rp = h + row * DM;
    int c0 = lane * 8;
    float v[8];
    #pragma unroll
    for (int i = 0; i < 8; ++i) v[i] = rp[c0 + i];
    float s = 0.f;
    #pragma unroll
    for (int i = 0; i < 8; ++i) s += v[i];
    #pragma unroll
    for (int o = 16; o > 0; o >>= 1) s += __shfl_xor(s, o, 32);
    float mu = s * (1.f / DM);
    float q = 0.f;
    #pragma unroll
    for (int i = 0; i < 8; ++i) { float d = v[i] - mu; q += d * d; }
    #pragma unroll
    for (int o = 16; o > 0; o >>= 1) q += __shfl_xor(q, o, 32);
    float inv = rsqrtf(q * (1.f / DM) + 1e-5f);
    bf16* op = out + row * DM;
    #pragma unroll
    for (int i = 0; i < 8; ++i)
        op[c0 + i] = (bf16)((v[i] - mu) * inv * g[c0 + i] + bta[c0 + i]);
}

// depthwise causal conv(4) + bias + SiLU; writes f32 (scan) and bf16 (x_proj GEMM)
__global__ void conv_silu_kernel(const float* __restrict__ xz, const float* __restrict__ cw,
                                 const float* __restrict__ cb, float* __restrict__ xc,
                                 bf16* __restrict__ xcb) {
    int i = blockIdx.x * blockDim.x + threadIdx.x;   // BL*DI
    if (i >= BL * DI) return;
    int d = i & (DI - 1);
    int bt = i >> 9;
    int t = bt & (SEQ - 1);
    size_t bb = (size_t)(bt >> 10);
    float s = cb[d];
    #pragma unroll
    for (int j = 0; j < DC; ++j) {
        int tt = t - (DC - 1) + j;
        if (tt >= 0) s += cw[d * DC + j] * xz[(bb * SEQ + tt) * (2 * DI) + d];
    }
    float v = siluf(s);
    xc[i]  = v;
    xcb[i] = (bf16)v;
}

// x_dbl[:, :16] -> bf16, K padded 16->32 with zeros (for dt_proj WMMA GEMM)
__global__ void dtprep_kernel(const float* __restrict__ xdbl, bf16* __restrict__ dst) {
    int i = blockIdx.x * blockDim.x + threadIdx.x;   // BL*32
    if (i >= BL * 32) return;
    int col = i & 31;
    size_t row = (size_t)(i >> 5);
    dst[i] = (bf16)(col < DTR ? xdbl[row * XD + col] : 0.f);
}

// selective scan: one thread per (b,d), 16 f32 states in registers,
// B_t/C_t (32 contiguous floats of x_dbl) broadcast via LDS. y overwrites xc.
__global__ void scan_kernel(const float* __restrict__ dtraw, const float* __restrict__ xdbl,
                            const float* __restrict__ dtb, const float* __restrict__ Alog,
                            const float* __restrict__ Dsk, float* __restrict__ xc_y) {
    const int b = blockIdx.x;
    const int d = blockIdx.y * 256 + threadIdx.x;
    __shared__ float sBC[32];
    float st[DS], Av[DS];
    #pragma unroll
    for (int n = 0; n < DS; ++n) { st[n] = 0.f; Av[n] = -__expf(Alog[d * DS + n]); }
    const float bias = dtb[d];
    const float dsk  = Dsk[d];
    for (int t = 0; t < SEQ; ++t) {
        size_t tok = (size_t)b * SEQ + t;
        if (threadIdx.x < 32) sBC[threadIdx.x] = xdbl[tok * XD + DTR + threadIdx.x];
        __syncthreads();
        size_t off = tok * DI + d;
        float dtv = softplusf(dtraw[off] + bias);
        float x   = xc_y[off];
        float dtx = dtv * x;
        float y = 0.f;
        #pragma unroll
        for (int n = 0; n < DS; ++n) {
            st[n] = st[n] * __expf(dtv * Av[n]) + dtx * sBC[n];
            y += st[n] * sBC[DS + n];
        }
        xc_y[off] = y + x * dsk;
        __syncthreads();
    }
}

// g = y * silu(z) -> bf16 (feeds out_proj WMMA GEMM)
__global__ void gate_kernel(const float* __restrict__ y, const float* __restrict__ xz,
                            bf16* __restrict__ g) {
    int i = blockIdx.x * blockDim.x + threadIdx.x;   // BL*DI
    if (i >= BL * DI) return;
    int d = i & (DI - 1);
    size_t bt = (size_t)(i >> 9);
    float z = xz[bt * (2 * DI) + DI + d];
    g[i] = (bf16)(y[i] * siluf(z));
}

__global__ void maskrow_kernel(const float* __restrict__ mask, float* __restrict__ mrow) {
    int i = blockIdx.x * blockDim.x + threadIdx.x;   // BL
    if (i >= BL) return;
    float s = 0.f;
    const float* p = mask + (size_t)i * NIN;
    #pragma unroll 8
    for (int k = 0; k < NIN; ++k) s += p[k];
    mrow[i] = s * (1.f / NIN);
}

__global__ void pool_kernel(const float* __restrict__ h, const float* __restrict__ mrow,
                            float* __restrict__ pooled) {
    int b = blockIdx.x, c = threadIdx.x;             // 256 threads
    __shared__ float sm[DM];
    float ms = 0.f;
    for (int t = c; t < SEQ; t += DM) ms += mrow[b * SEQ + t];
    sm[c] = ms; __syncthreads();
    for (int s = DM / 2; s > 0; s >>= 1) { if (c < s) sm[c] += sm[c + s]; __syncthreads(); }
    float msum = sm[0];
    float acc = 0.f;
    for (int t = 0; t < SEQ; ++t)
        acc += h[((size_t)b * SEQ + t) * DM + c] * mrow[b * SEQ + t];
    pooled[b * DM + c] = acc / (msum + 1e-8f);
}

// fc1(256->128)+exact GELU, fc2(128->1). One block (128 threads) per batch row.
__global__ void head_kernel(const float* __restrict__ pooled,
                            const float* __restrict__ w1, const float* __restrict__ b1,
                            const float* __restrict__ w2, const float* __restrict__ b2,
                            float* __restrict__ out) {
    int b = blockIdx.x, j = threadIdx.x;             // 128 threads
    float a = b1[j];
    const float* p = pooled + (size_t)b * DM;
    #pragma unroll 8
    for (int k = 0; k < DM; ++k) a += p[k] * w1[j * DM + k];
    float ge = 0.5f * a * (1.f + erff(a * 0.70710678118654752f));
    __shared__ float s[128];
    s[j] = ge * w2[j]; __syncthreads();
    for (int st = 64; st > 0; st >>= 1) { if (j < st) s[j] += s[j + st]; __syncthreads(); }
    if (j == 0) out[b] = s[0] + b2[0];
}

// ---------------------------------------------------------------------------
extern "C" void kernel_launch(void* const* d_in, const int* in_sizes, int n_in,
                              void* d_out, int out_size, void* d_ws, size_t ws_size,
                              hipStream_t stream) {
    (void)in_sizes; (void)n_in; (void)out_size; (void)ws_size;
    const float* x    = (const float*)d_in[0];
    const float* mask = (const float*)d_in[1];
    const float* ipw  = (const float*)d_in[2];
    const float* ipb  = (const float*)d_in[3];
    const float* inw  = (const float*)d_in[4];
    const float* cw   = (const float*)d_in[5];
    const float* cb   = (const float*)d_in[6];
    const float* xpw  = (const float*)d_in[7];
    const float* dtw  = (const float*)d_in[8];
    const float* dtb  = (const float*)d_in[9];
    const float* alog = (const float*)d_in[10];
    const float* dsk  = (const float*)d_in[11];
    const float* ow   = (const float*)d_in[12];
    const float* lng  = (const float*)d_in[13];
    const float* lnb  = (const float*)d_in[14];
    const float* f1w  = (const float*)d_in[15];
    const float* f1b  = (const float*)d_in[16];
    const float* f2w  = (const float*)d_in[17];
    const float* f2b  = (const float*)d_in[18];
    float* out = (float*)d_out;

    // ---- workspace carve-up (~191 MB; whole working set ~ L2-resident) ----
    char* ws = (char*)d_ws;
    size_t cur = 0;
    auto carve = [&](size_t bytes) -> char* {
        char* p = ws + cur; cur += (bytes + 255) & ~(size_t)255; return p;
    };
    float* h      = (float*)carve((size_t)BL * DM * 4);        // residual stream
    float* xz     = (float*)carve((size_t)BL * 2 * DI * 4);    // in_proj out
    float* xc     = (float*)carve((size_t)BL * DI * 4);        // conv out, then y
    float* xdbl   = (float*)carve((size_t)BL * XD * 4);        // x_proj out
    float* dtraw  = (float*)carve((size_t)BL * DI * 4);        // dt_proj out
    float* mrow   = (float*)carve((size_t)BL * 4);
    float* pooled = (float*)carve((size_t)BATCH * DM * 4);
    bf16* lnb16   = (bf16*)carve((size_t)BL * DM * 2);
    bf16* xcb16   = (bf16*)carve((size_t)BL * DI * 2);
    bf16* dtin16  = (bf16*)carve((size_t)BL * 32 * 2);
    bf16* g16     = (bf16*)carve((size_t)BL * DI * 2);
    bf16* w_in    = (bf16*)carve((size_t)NL * 2 * DI * DM * 2);
    bf16* w_xp    = (bf16*)carve((size_t)NL * XD * DI * 2);
    bf16* w_dt    = (bf16*)carve((size_t)NL * DI * 32 * 2);
    bf16* w_out   = (bf16*)carve((size_t)NL * DM * DI * 2);

    auto g1 = [](long long n, int b) { return (unsigned)((n + b - 1) / b); };

    // (N/16) % NBLK == 0 for every launch below: 64,32,16 -> NBLK=4; 3 -> NBLK=3
    auto launch_gemm = [&](const bf16* Aact, const bf16* Wp, float* Cout,
                           int N, int K, int accum) {
        int nTiles = N / 16;
        if (nTiles % 4 == 0) {
            dim3 grid(nTiles / 4, (BL / 16) / 8);
            size_t shb = (size_t)4 * 16 * K * 2;
            gemm_bf16_wmma<4><<<grid, 256, shb, stream>>>(Aact, Wp, Cout, BL, N, K, accum);
        } else {
            dim3 grid(nTiles / 3, (BL / 16) / 8);
            size_t shb = (size_t)3 * 16 * K * 2;
            gemm_bf16_wmma<3><<<grid, 256, shb, stream>>>(Aact, Wp, Cout, BL, N, K, accum);
        }
    };

    // ---- weight conversion to bf16 (cheap, deterministic each call) ----
    { int n = NL * 2 * DI * DM; cvt_bf16_kernel<<<g1(n,256),256,0,stream>>>(inw, w_in, n); }
    { int n = NL * XD * DI;     cvt_bf16_kernel<<<g1(n,256),256,0,stream>>>(xpw, w_xp, n); }
    { int n = NL * DM * DI;     cvt_bf16_kernel<<<g1(n,256),256,0,stream>>>(ow,  w_out, n); }
    dtw_pad_kernel<<<g1(NL*DI*32,256),256,0,stream>>>(dtw, w_dt);

    // ---- input projection + mask row means ----
    inproj_kernel<<<g1((long long)BL*DM,256),256,0,stream>>>(x, ipw, ipb, h);
    maskrow_kernel<<<g1(BL,256),256,0,stream>>>(mask, mrow);

    // ---- 4 Mamba layers ----
    for (int l = 0; l < NL; ++l) {
        ln_bf16_kernel<<<BL/8,256,0,stream>>>(h, lng + l*DM, lnb + l*DM, lnb16);

        // xz = ln @ in_proj_w.T           M=16384 K=256 N=1024
        launch_gemm(lnb16, w_in + (size_t)l*2*DI*DM, xz, 2*DI, DM, 0);

        conv_silu_kernel<<<g1((long long)BL*DI,256),256,0,stream>>>(
            xz, cw + l*DI*DC, cb + l*DI, xc, xcb16);

        // x_dbl = xc @ x_proj_w.T         M=16384 K=512 N=48
        launch_gemm(xcb16, w_xp + (size_t)l*XD*DI, xdbl, XD, DI, 0);

        dtprep_kernel<<<g1(BL*32,256),256,0,stream>>>(xdbl, dtin16);

        // dt_raw = dt_in @ dt_proj_w.T    M=16384 K=32(pad) N=512
        launch_gemm(dtin16, w_dt + (size_t)l*DI*32, dtraw, DI, 32, 0);

        // sequential selective scan (y overwrites xc)
        scan_kernel<<<dim3(BATCH,2),256,0,stream>>>(dtraw, xdbl, dtb + l*DI,
                                                    alog + (size_t)l*DI*DS, dsk + l*DI, xc);

        gate_kernel<<<g1((long long)BL*DI,256),256,0,stream>>>(xc, xz, g16);

        // h += g @ out_proj_w.T           M=16384 K=512 N=256  (residual accumulate)
        launch_gemm(g16, w_out + (size_t)l*DM*DI, h, DM, DI, 1);
    }

    // ---- masked mean pool + GELU MLP head ----
    pool_kernel<<<BATCH,DM,0,stream>>>(h, mrow, pooled);
    head_kernel<<<BATCH,128,0,stream>>>(pooled, f1w, f1b, f2w, f2b, out);
}
